// SCAESuite_2748779069914
// MI455X (gfx1250) — matrix operand order repaired and checked by reading, not compiled, not run on hardware
//
#include <hip/hip_runtime.h>

// ---------------------------------------------------------------------------
// Types
// ---------------------------------------------------------------------------
typedef __attribute__((ext_vector_type(8)))  __bf16 v8bf;
typedef __attribute__((ext_vector_type(16))) __bf16 v16bf;
typedef __attribute__((ext_vector_type(8)))  float  v8f;
typedef __attribute__((ext_vector_type(4)))  unsigned v4u;
typedef __attribute__((ext_vector_type(8)))  unsigned v8u;

__device__ __forceinline__ __bf16 to_bf16(float f) {
  union { float f; unsigned u; } a; a.f = f;
  unsigned r = a.u + 0x7FFFu + ((a.u >> 16) & 1u);   // round-to-nearest-even
  union { unsigned short s; __bf16 h; } o; o.s = (unsigned short)(r >> 16);
  return o.h;
}

__device__ __forceinline__ v8bf ld_v8bf(const __bf16* p) {
  v8bf r; __builtin_memcpy(&r, p, sizeof(r)); return r;
}
__device__ __forceinline__ v16bf ld_v16bf(const __bf16* p) {
  v16bf r; __builtin_memcpy(&r, p, sizeof(r)); return r;
}

// ---------------------------------------------------------------------------
// Prep kernels
// ---------------------------------------------------------------------------
__global__ __launch_bounds__(256) void k_prep_act(
    const float* __restrict__ xu, const float* __restrict__ x0,
    const float* __restrict__ b_dec_up,
    __bf16* __restrict__ actUp, __bf16* __restrict__ act0,
    int total, int D) {
  int i = blockIdx.x * 256 + threadIdx.x;
  if (i < total) {
    int d = i % D;
    actUp[i] = to_bf16(xu[i] - b_dec_up[d]);
    act0[i]  = to_bf16(x0[i]);
  }
}

__global__ __launch_bounds__(256) void k_f32_to_bf16(
    const float* __restrict__ in, __bf16* __restrict__ out, int total) {
  int i = blockIdx.x * 256 + threadIdx.x;
  if (i < total) out[i] = to_bf16(in[i]);
}

// in: [R, Cn] row-major  ->  out: [Cn, R] row-major
__global__ __launch_bounds__(256) void k_transpose(
    const float* __restrict__ in, float* __restrict__ out, int R, int Cn) {
  __shared__ float t[32][33];
  int bx = blockIdx.x * 32, by = blockIdx.y * 32;
  int x = bx + threadIdx.x;
  for (int k = 0; k < 32; k += 8) {
    int y = by + threadIdx.y + k;
    if (y < R && x < Cn) t[threadIdx.y + k][threadIdx.x] = in[(size_t)y * Cn + x];
  }
  __syncthreads();
  int x2 = by + threadIdx.x;
  for (int k = 0; k < 32; k += 8) {
    int y2 = bx + threadIdx.y + k;
    if (y2 < Cn && x2 < R) out[(size_t)y2 * R + x2] = t[threadIdx.x][threadIdx.y + k];
  }
}

// c1[i] = <W[i,:], b_up>, c2[i] = <W[i,:], b_dn>   (one wave per row)
__global__ __launch_bounds__(32) void k_cvec(
    const float* __restrict__ W, const float* __restrict__ b_up,
    const float* __restrict__ b_dn, float* __restrict__ c1,
    float* __restrict__ c2, int D) {
  int i = blockIdx.x, lane = threadIdx.x;
  const float* w = W + (size_t)i * D;
  float a = 0.f, b = 0.f;
  for (int d = lane; d < D; d += 32) { float wv = w[d]; a += wv * b_up[d]; b += wv * b_dn[d]; }
  for (int m = 16; m; m >>= 1) { a += __shfl_xor(a, m, 32); b += __shfl_xor(b, m, 32); }
  if (lane == 0) { c1[i] = a; c2[i] = b; }
}

// ---------------------------------------------------------------------------
// WMMA GEMM:  C[M,N] = A[M,K](bf16, row-major) x B[N,K](bf16, row-major, used
// as K x N)  with optional bias+relu epilogue.
// Workgroup tile 128x128, 8 waves (4 in M x 2 in N), each wave 32x64 =
// 2x4 v_wmma_f32_16x16x32_bf16 tiles.  K stepped by 32 (= WMMA K).
// Data movement (CDNA5):
//   A tile:  GLOBAL_LOAD_ASYNC_TO_LDS_B128 (ASYNCcnt), per-lane DMA to LDS
//   B tile:  TENSOR_LOAD_TO_LDS (TDM, TENSORcnt), 2D descriptor with HW
//            padding (16 dwords data + 4 dwords pad = the 80B LDS row stride)
// ---------------------------------------------------------------------------
#define GTM 128
#define GTN 128
#define GTK 32
#define LDA 40   // padded LDS row stride (halves): 80B -> conflict-free frags
#define LDB 40

__global__ __launch_bounds__(256, 2) void k_gemm_bf16_wmma(
    const __bf16* __restrict__ A, const __bf16* __restrict__ B,
    const float* __restrict__ bias, float* __restrict__ C,
    int M, int N, int K, int relu) {
  __shared__ __bf16 sA[GTM * LDA];
  __shared__ __bf16 sB[GTN * LDB];

  const int tid  = threadIdx.x;
  const int lane = tid & 31;
  const int wave = tid >> 5;
  const int wm   = wave & 3;   // M sub-tile: 32 rows
  const int wn   = wave >> 2;  // N sub-tile: 64 cols
  const int bm   = blockIdx.y * GTM;
  const int bn   = blockIdx.x * GTN;
  const int r16  = lane & 15;
  const int g    = lane >> 4;

  v8f acc[2][4];
#pragma unroll
  for (int mi = 0; mi < 2; ++mi)
#pragma unroll
    for (int ni = 0; ni < 4; ++ni)
#pragma unroll
      for (int e = 0; e < 8; ++e) acc[mi][ni][e] = 0.f;

  const int lr = tid >> 1;          // 0..127 A-tile row being copied
  const int lc = (tid & 1) * 16;    // 0 or 16 (halves)

  for (int k0 = 0; k0 < K; k0 += GTK) {
    {  // A tile [128 rows][32 halves] -> LDS via async DMA (no VGPR roundtrip)
      const __bf16* src = A + (size_t)(bm + lr) * K + k0 + lc;
      unsigned lds_off = (unsigned)(unsigned long long)(sA + lr * LDA + lc);
      unsigned long long ga = (unsigned long long)src;
      asm volatile("global_load_async_to_lds_b128 %0, %1, off"
                   :: "v"(lds_off), "v"(ga) : "memory");
      asm volatile("global_load_async_to_lds_b128 %0, %1, off offset:16"
                   :: "v"(lds_off), "v"(ga) : "memory");
      if (k0 + GTK < K) __builtin_prefetch(src + GTK, 0, 1);  // global_prefetch_b8
    }
    if (wave == 0) {  // B tile via Tensor Data Mover (one issue per workgroup)
      // D# group0: count=1, lds_addr, 57b global addr, type=2 ("image")
      unsigned long long ga = (unsigned long long)(B + (size_t)bn * K + k0);
      unsigned sbase = __builtin_amdgcn_readfirstlane((unsigned)(unsigned long long)sB);
      unsigned galo  = __builtin_amdgcn_readfirstlane((unsigned)(ga & 0xFFFFFFFFull));
      unsigned gahi  = __builtin_amdgcn_readfirstlane((unsigned)(ga >> 32));
      v4u g0;
      g0[0] = 1u;                    // count=1, user descriptor
      g0[1] = sbase;                 // lds_addr
      g0[2] = galo;                  // global_addr[31:0]
      g0[3] = gahi | (2u << 30);     // global_addr[56:32] | type=2
      // D# group1: data_size=4B units; row = 16 dwords, pad 4 dwords after
      // every 16 dwords (pad_interval=3, pad_amount=3) -> 80B LDS row stride.
      unsigned kd = (unsigned)K >> 1;          // row length in dwords
      v8u g1;
      g1[0] = (2u << 16)             // data_size = 4B
            | (1u << 20)             // pad_enable
            | (3u << 22)             // pad_interval: 16 dwords
            | (3u << 25);            // pad_amount:   4 dwords
      g1[1] = (kd & 0xFFFFu) << 16;                      // tensor_dim0[15:0]
      g1[2] = (kd >> 16) | (((unsigned)N & 0xFFFFu) << 16); // dim0 hi | tensor_dim1 lo
      g1[3] = ((unsigned)N >> 16) | (16u << 16);         // tensor_dim1 hi | tile_dim0=16
      g1[4] = (unsigned)GTN;                             // tile_dim1 = 128 rows
      g1[5] = kd;                                        // tensor_dim0_stride[31:0]
      g1[6] = 0u;                                        // stride hi, dim1_stride lo
      g1[7] = 0u;
      asm volatile("tensor_load_to_lds %0, %1" :: "s"(g0), "s"(g1) : "memory");
      __builtin_amdgcn_s_wait_tensorcnt(0);
    }
    asm volatile("s_wait_asynccnt 0x0" ::: "memory");
    __syncthreads();

    // A fragments: lane=M row; elements 0..15 hold contiguous K=16g..16g+15
    v16bf af[2];
#pragma unroll
    for (int mi = 0; mi < 2; ++mi) {
      const __bf16* pa = sA + (wm * 32 + mi * 16 + r16) * LDA;
      v8bf lo = ld_v8bf(pa + 16 * g);
      v8bf hi = ld_v8bf(pa + 16 * g + 8);
      af[mi] = __builtin_shufflevector(lo, hi, 0, 1, 2, 3, 4, 5, 6, 7,
                                       8, 9, 10, 11, 12, 13, 14, 15);
    }
    // B fragments: lane=N col; elements 0..15 hold contiguous K=16g..16g+15
    v16bf bfr[4];
#pragma unroll
    for (int ni = 0; ni < 4; ++ni) {
      const __bf16* pb = sB + (wn * 64 + ni * 16 + r16) * LDB + 16 * g;
      bfr[ni] = ld_v16bf(pb);
    }

#pragma unroll
    for (int mi = 0; mi < 2; ++mi)
#pragma unroll
      for (int ni = 0; ni < 4; ++ni)
        acc[mi][ni] = __builtin_amdgcn_wmma_f32_16x16x32_bf16(
            false, af[mi], false, bfr[ni], (short)0, acc[mi][ni], false, false);
    __syncthreads();
  }

  // Epilogue.  C/D layout: VGPR r, lane L -> M = r + 8*(L/16), N = L%16
#pragma unroll
  for (int mi = 0; mi < 2; ++mi)
#pragma unroll
    for (int ni = 0; ni < 4; ++ni) {
      int col = bn + wn * 64 + ni * 16 + r16;
      float bval = bias ? bias[col] : 0.f;
#pragma unroll
      for (int r = 0; r < 8; ++r) {
        int row = bm + wm * 32 + mi * 16 + 8 * g + r;
        float v = acc[mi][ni][r] + bval;
        if (relu) v = fmaxf(v, 0.f);
        C[(size_t)row * N + col] = v;
      }
    }
}

// ---------------------------------------------------------------------------
// Masked virtual weights: V[i,c] = <W_enc_down[i,:], WdecUpT[conn[i,c],:]>,
// zeroed for duplicate conn entries (matches the boolean-mask semantics).
// One workgroup per down-feature i; one wave per connection c.
// ---------------------------------------------------------------------------
__global__ __launch_bounds__(256) void k_virtual(
    const float* __restrict__ WencDown, const float* __restrict__ WdecUpT,
    const int* __restrict__ conn, float* __restrict__ V, int D, int C) {
  __shared__ float wrow[768];
  int i = blockIdx.x, tid = threadIdx.x, lane = tid & 31, wave = tid >> 5;
  for (int d = tid; d < D; d += 256) wrow[d] = WencDown[(size_t)i * D + d];
  __syncthreads();
  for (int c = wave; c < C; c += 8) {
    int j = conn[i * C + c];
    int dupl = 0;
    for (int c2 = lane; c2 < c; c2 += 32)
      if (conn[i * C + c2] == j) dupl = 1;
    dupl = (__ballot(dupl) != 0);
    float acc = 0.f;
    if (!dupl) {
      const float* wr = WdecUpT + (size_t)j * D;
      for (int d = lane; d < D; d += 32) acc += wrow[d] * wr[d];
    }
    for (int m = 16; m; m >>= 1) acc += __shfl_xor(acc, m, 32);
    if (lane == 0) V[i * C + c] = dupl ? 0.f : acc;
  }
}

// ---------------------------------------------------------------------------
// Row-wise top-K (K=64) by iterative argmax over an LDS-staged row.
// Ties resolved toward the lowest index (jax.lax.top_k ordering).
// Optionally emits the dense (zero-filled) feature row.
// ---------------------------------------------------------------------------
__global__ __launch_bounds__(256) void k_topk(
    const float* __restrict__ X, float* __restrict__ vals,
    int* __restrict__ idxs, float* __restrict__ dense,
    int F, int K, int writeDense) {
  __shared__ float srow[8192];
  __shared__ unsigned sel[256];
  __shared__ float redV[256];
  __shared__ int   redI[256];
  int n = blockIdx.x, tid = threadIdx.x;
  const float* row = X + (size_t)n * F;
  for (int j = tid; j < F; j += 256) srow[j] = row[j];
  sel[tid] = 0u;
  __syncthreads();
  for (int it = 0; it < K; ++it) {
    float bv = -3.402823466e38f; int bi = F;
    for (int j = tid; j < F; j += 256) {
      if ((sel[j >> 5] >> (j & 31)) & 1u) continue;
      float v = srow[j];
      if (v > bv || (v == bv && j < bi)) { bv = v; bi = j; }
    }
    redV[tid] = bv; redI[tid] = bi;
    __syncthreads();
    for (int s = 128; s > 0; s >>= 1) {
      if (tid < s) {
        float v2 = redV[tid + s]; int i2 = redI[tid + s];
        if (v2 > redV[tid] || (v2 == redV[tid] && i2 < redI[tid])) {
          redV[tid] = v2; redI[tid] = i2;
        }
      }
      __syncthreads();
    }
    if (tid == 0) {
      int w = redI[0];
      vals[n * K + it] = redV[0];
      idxs[n * K + it] = w;
      sel[w >> 5] |= (1u << (w & 31));
    }
    __syncthreads();
  }
  if (writeDense) {
    float* drow = dense + (size_t)n * F;
    for (int j = tid; j < F; j += 256)
      drow[j] = ((sel[j >> 5] >> (j & 31)) & 1u) ? srow[j] : 0.f;
  }
}

// ---------------------------------------------------------------------------
// Pruned contributions + epilogue, fused and in-place on the down-GEMM output:
// out[n,i] = (gemm[n,i] + sum_c feat[n,conn[i,c]]*V[i,c] + c1[i]) / ln[n]
//            + b_enc_down[i] - c2[i]
// ---------------------------------------------------------------------------
__global__ __launch_bounds__(256) void k_pruned_assemble(
    float* __restrict__ gemmD, const float* __restrict__ featDense,
    const float* __restrict__ V, const int* __restrict__ conn,
    const float* __restrict__ c1, const float* __restrict__ c2,
    const float* __restrict__ b_enc_down, const float* __restrict__ ln,
    int F, int C) {
  __shared__ float srow[8192];
  int n = blockIdx.x, tid = threadIdx.x, lane = tid & 31, wave = tid >> 5;
  const float* fr = featDense + (size_t)n * F;
  for (int j = tid; j < F; j += 256) srow[j] = fr[j];
  __syncthreads();
  float inv = 1.0f / ln[n];
  for (int i = wave; i < F; i += 8) {
    float acc = 0.f;
    for (int c = lane; c < C; c += 32) {
      int j = conn[i * C + c];
      acc += srow[j] * V[i * C + c];
    }
    for (int m = 16; m; m >>= 1) acc += __shfl_xor(acc, m, 32);
    if (lane == 0) {
      size_t o = (size_t)n * F + i;
      float t = gemmD[o] + acc + c1[i];
      gemmD[o] = t * inv + b_enc_down[i] - c2[i];
    }
  }
}

// ---------------------------------------------------------------------------
// Sparse decode: out[n,:] = b_dec + sum_k vals[n,k] * WdecT[idx[n,k], :]
// ---------------------------------------------------------------------------
__global__ __launch_bounds__(256) void k_recon(
    const float* __restrict__ vals, const int* __restrict__ idxs,
    const float* __restrict__ WdecT, const float* __restrict__ b_dec,
    float* __restrict__ out, int D, int K) {
  __shared__ float sv[64];
  __shared__ int   si[64];
  int n = blockIdx.x, tid = threadIdx.x;
  if (tid < K) { sv[tid] = vals[n * K + tid]; si[tid] = idxs[n * K + tid]; }
  __syncthreads();
  for (int d = tid; d < D; d += 256) {
    float a = b_dec[d];
    for (int k = 0; k < K; ++k) a += sv[k] * WdecT[(size_t)si[k] * D + d];
    out[(size_t)n * D + d] = a;
  }
}

// ---------------------------------------------------------------------------
// Host launcher
// ---------------------------------------------------------------------------
extern "C" void kernel_launch(void* const* d_in, const int* in_sizes, int n_in,
                              void* d_out, int out_size, void* d_ws, size_t ws_size,
                              hipStream_t stream) {
  const float* x0      = (const float*)d_in[0];
  const float* xu      = (const float*)d_in[1];
  const float* ln      = (const float*)d_in[2];
  const float* WencUp  = (const float*)d_in[3];
  const float* bEncUp  = (const float*)d_in[4];
  const float* WdecUp  = (const float*)d_in[5];   // [D, F]
  const float* bDecUp  = (const float*)d_in[6];
  const float* WencDn  = (const float*)d_in[7];
  const float* bEncDn  = (const float*)d_in[8];
  const float* WdecDn  = (const float*)d_in[9];   // [D, F]
  const float* bDecDn  = (const float*)d_in[10];
  const int*   conn    = (const int*)d_in[11];
  float* out = (float*)d_out;

  const int N  = in_sizes[2];              // B*S tokens (1024)
  const int D  = in_sizes[6];              // 768
  const int F  = in_sizes[4];              // 8192
  const int C  = in_sizes[11] / F;         // 128
  const int K  = 64;

  char* ws = (char*)d_ws;
  auto alloc = [&](size_t bytes) -> void* {
    void* p = (void*)ws;
    ws += (bytes + 255) & ~(size_t)255;
    return p;
  };
  __bf16* actUp    = (__bf16*)alloc((size_t)N * D * 2);
  __bf16* act0     = (__bf16*)alloc((size_t)N * D * 2);
  __bf16* WencUpB  = (__bf16*)alloc((size_t)F * D * 2);
  __bf16* WencDnB  = (__bf16*)alloc((size_t)F * D * 2);
  float*  WdecUpT  = (float*)alloc((size_t)F * D * 4);
  float*  WdecDnT  = (float*)alloc((size_t)F * D * 4);
  float*  c1       = (float*)alloc((size_t)F * 4);
  float*  c2       = (float*)alloc((size_t)F * 4);
  float*  postUp   = (float*)alloc((size_t)N * F * 4);  // also reused as scratch
  float*  featDen  = (float*)alloc((size_t)N * F * 4);
  float*  gemmD    = (float*)alloc((size_t)N * F * 4);  // becomes approx in place
  float*  V        = (float*)alloc((size_t)F * C * 4);
  float*  upVals   = (float*)alloc((size_t)N * K * 4);
  int*    upIdx    = (int*)alloc((size_t)N * K * 4);
  float*  dnVals   = (float*)alloc((size_t)N * K * 4);
  int*    dnIdx    = (int*)alloc((size_t)N * K * 4);

  // --- prep ---
  {
    int total = N * D;
    k_prep_act<<<(total + 255) / 256, 256, 0, stream>>>(xu, x0, bDecUp, actUp, act0, total, D);
  }
  {
    int total = F * D;
    k_f32_to_bf16<<<(total + 255) / 256, 256, 0, stream>>>(WencUp, WencUpB, total);
    k_f32_to_bf16<<<(total + 255) / 256, 256, 0, stream>>>(WencDn, WencDnB, total);
  }
  {
    dim3 blk(32, 8), grd(F / 32, D / 32);
    k_transpose<<<grd, blk, 0, stream>>>(WdecUp, WdecUpT, D, F);
    k_transpose<<<grd, blk, 0, stream>>>(WdecDn, WdecDnT, D, F);
  }
  k_cvec<<<F, 32, 0, stream>>>(WencDn, bDecUp, bDecDn, c1, c2, D);

  // --- upstream SAE ---
  {
    dim3 grd(F / GTN, N / GTM);
    k_gemm_bf16_wmma<<<grd, 256, 0, stream>>>(actUp, WencUpB, bEncUp, postUp,
                                              N, F, D, /*relu=*/1);
  }
  k_topk<<<N, 256, 0, stream>>>(postUp, upVals, upIdx, featDen, F, K, /*dense=*/1);
  k_recon<<<N, 256, 0, stream>>>(upVals, upIdx, WdecUpT, bDecUp, out, D, K);

  // --- pruned downstream path ---
  k_virtual<<<F, 256, 0, stream>>>(WencDn, WdecUpT, conn, V, D, C);
  {
    dim3 grd(F / GTN, N / GTM);
    k_gemm_bf16_wmma<<<grd, 256, 0, stream>>>(act0, WencDnB, nullptr, gemmD,
                                              N, F, D, /*relu=*/0);
  }
  k_pruned_assemble<<<N, 256, 0, stream>>>(gemmD, featDen, V, conn, c1, c2,
                                           bEncDn, ln, F, C);
  k_topk<<<N, 256, 0, stream>>>(gemmD, dnVals, dnIdx, postUp /*unused*/, F, K, /*dense=*/0);
  k_recon<<<N, 256, 0, stream>>>(dnVals, dnIdx, WdecDnT, bDecDn,
                                 out + (size_t)N * D, D, K);
}
